// SparseAttention_58583353918008
// MI455X (gfx1250) — matrix operand order repaired and checked by reading, not compiled, not run on hardware
//
#include <hip/hip_runtime.h>

// ---------------------------------------------------------------------------
// Sparse attention for MI455X (gfx1250): bf16 WMMA (16x16x32) GEMM pipeline
// + Tensor Data Mover (tensor_load_to_lds) staging for the attention tiles.
// f32 inputs are converted to bf16 while staging LDS tiles; accumulation f32.
// ---------------------------------------------------------------------------

#define HIDDEN   2048
#define HEADS    16
#define HEAD_DIM 128
#define TOPK     256
#define NSEQ     2048

typedef __bf16 bf16;
typedef __attribute__((ext_vector_type(16))) __bf16 bf16x16;
typedef __attribute__((ext_vector_type(8)))  float  f32x8;
typedef __attribute__((ext_vector_type(4)))  unsigned int u32x4;
typedef __attribute__((ext_vector_type(8)))  int i32x8;
typedef __attribute__((ext_vector_type(4)))  int i32x4;

__device__ __forceinline__ f32x8 zero8() {
  f32x8 z;
#pragma unroll
  for (int i = 0; i < 8; ++i) z[i] = 0.0f;
  return z;
}

__device__ __forceinline__ f32x8 wmma_bf16(bf16x16 a, bf16x16 b, f32x8 c) {
  // D = A(16x32 bf16) x B(32x16 bf16) + C(16x16 f32)
  return __builtin_amdgcn_wmma_f32_16x16x32_bf16(
      /*neg_a=*/false, a, /*neg_b=*/false, b,
      /*c_mod=*/(short)0, c, /*reuse_a=*/false, /*reuse_b=*/false);
}

union FragU { uint4 u[2]; bf16x16 v; };

// A fragment (16x32, row-major tile in LDS, stride ld elements).
// Layout: lanes 0-15 M=0..15 hold K 0..15 (contiguous);
// lanes 16-31 hold K 16..31.
__device__ __forceinline__ bf16x16 ldsA_frag(const bf16* base, int ld) {
  const int lane = threadIdx.x & 31;
  const int h = lane >> 4, m = lane & 15;
  const bf16* p = base + m * ld + h * 16;
  FragU f;
  f.u[0] = *reinterpret_cast<const uint4*>(p);
  f.u[1] = *reinterpret_cast<const uint4*>(p + 8);
  return f.v;
}

// B fragment (32x16) from a TRANSPOSED tile in LDS: element (k,n) at
// base[n*ldk + k].  Per lane: col n = lane&15, K = (lane>>4)*16 .. +15,
// i.e. 16 contiguous bf16 per lane.
__device__ __forceinline__ bf16x16 ldsBT_frag(const bf16* base, int ldk) {
  const int lane = threadIdx.x & 31;
  const int h = lane >> 4, n = lane & 15;
  const bf16* p = base + n * ldk + h * 16;
  FragU f;
  f.u[0] = *reinterpret_cast<const uint4*>(p);
  f.u[1] = *reinterpret_cast<const uint4*>(p + 8);
  return f.v;
}

__device__ __forceinline__ unsigned pack2bf(float a, float b) {
  union { bf16 h[2]; unsigned u; } x;
  x.h[0] = (bf16)a; x.h[1] = (bf16)b;
  return x.u;
}

// ---------------------------------------------------------------------------
// Tensor Data Mover: build a D# per CDNA5 ISA and issue
// TENSOR_LOAD_TO_LDS.  2D tile (tile0 x tile1) of 2-byte elements, row stride
// stride0 elements; LDS destination packed contiguously.
// This toolchain exposes the 6-arg builtin:
//   (u32x4 g0, i32x8 g1, i32x4 g2, i32x4 g3, i32x8 extra, i32 cpol)
// The first four args carry the ISA's four D# SGPR groups; extra is zero.
// ---------------------------------------------------------------------------
#if __has_builtin(__builtin_amdgcn_tensor_load_to_lds) && \
    __has_builtin(__builtin_amdgcn_s_wait_tensorcnt)
#define USE_TDM 1
#else
#define USE_TDM 0
#endif

#if USE_TDM
__device__ __forceinline__ void tdm_load_2d_b16(unsigned lds_byte_addr,
                                                const void* gptr,
                                                unsigned tile0, unsigned tile1,
                                                unsigned td0, unsigned td1,
                                                unsigned long long stride0) {
  const unsigned long long ga = (unsigned long long)(size_t)gptr;
  u32x4 g0;
  g0[0] = 1u;                                   // count=1, user descriptor
  g0[1] = lds_byte_addr;                        // lds_addr [63:32]
  g0[2] = (unsigned)ga;                         // global_addr [95:64]
  g0[3] = (unsigned)((ga >> 32) & 0x01ffffffu)  // global_addr [120:96]
          | (2u << 30);                         // type=2 ("image")
  i32x8 g1;
  g1[0] = (int)(1u << 16);                      // workgroup_mask=0, data_size=1 (2B)
  g1[1] = (int)((td0 & 0xffffu) << 16);         // tensor_dim0[15:0] @ [63:48]
  g1[2] = (int)((td0 >> 16) | ((td1 & 0xffffu) << 16));  // td0 hi | td1 lo
  g1[3] = (int)((td1 >> 16) | (tile0 << 16));   // td1 hi | tile_dim0
  g1[4] = (int)(tile1 & 0xffffu);               // tile_dim1 (tile_dim2=0)
  g1[5] = (int)(unsigned)(stride0 & 0xffffffffull);       // dim0 stride lo
  g1[6] = (int)(unsigned)((stride0 >> 32) & 0xffffull);   // stride hi | dim1 stride lo
  g1[7] = 0;
  i32x4 g2 = {0, 0, 0, 0};                      // <=2D: groups 2/3 unused
  i32x4 g3 = {0, 0, 0, 0};
  i32x8 gz = {0, 0, 0, 0, 0, 0, 0, 0};          // extra operand (zero-filled)
  __builtin_amdgcn_tensor_load_to_lds(g0, g1, g2, g3, gz, 0);
}
#endif

// ---------------------------------------------------------------------------
// Dense 2048x2048x2048 GEMM: out = A(f32) * W(f32) + bias, out bf16 or f32.
// Block tile 128x128, K-step 32, 8 waves: wave = (wm 0..3 rows of 32,
// wn 0..1 cols of 64); each wave owns 2x4 = 8 accum tiles of 16x16.
// ---------------------------------------------------------------------------
__global__ __launch_bounds__(256)
void gemm2048_bf16wmma(const float* __restrict__ A, const float* __restrict__ W,
                       const float* __restrict__ bias,
                       bf16* __restrict__ outB, float* __restrict__ outF) {
  __shared__ bf16 sA[128 * 32];   // [m][k]
  __shared__ bf16 sBT[128 * 32];  // [n][k] (transposed)

  const int tid  = threadIdx.x;
  const int lane = tid & 31;
  const int wave = tid >> 5;
  const int wm = wave & 3;
  const int wn = wave >> 2;
  const int m0 = blockIdx.x * 128;
  const int n0 = blockIdx.y * 128;

  f32x8 acc[2][4];
#pragma unroll
  for (int i = 0; i < 2; ++i)
#pragma unroll
    for (int j = 0; j < 4; ++j) acc[i][j] = zero8();

  for (int kt = 0; kt < HIDDEN / 32; ++kt) {
    const int k0 = kt * 32;
    // Stage A tile 128x32 (f32 -> bf16)
    for (int l = tid; l < 1024; l += 256) {
      const int row = l >> 3, c4 = l & 7;
      float4 v = *reinterpret_cast<const float4*>(
          A + (size_t)(m0 + row) * HIDDEN + k0 + c4 * 4);
      uint2 p;
      p.x = pack2bf(v.x, v.y);
      p.y = pack2bf(v.z, v.w);
      *reinterpret_cast<uint2*>(&sA[row * 32 + c4 * 4]) = p;
    }
    // Stage B tile 32x128 transposed (f32 -> bf16)
    for (int l = tid; l < 1024; l += 256) {
      const int kk = l >> 5, c4 = l & 31;
      float4 v = *reinterpret_cast<const float4*>(
          W + (size_t)(k0 + kk) * HIDDEN + n0 + c4 * 4);
      const int nb = c4 * 4;
      sBT[(nb + 0) * 32 + kk] = (bf16)v.x;
      sBT[(nb + 1) * 32 + kk] = (bf16)v.y;
      sBT[(nb + 2) * 32 + kk] = (bf16)v.z;
      sBT[(nb + 3) * 32 + kk] = (bf16)v.w;
    }
    if (kt + 1 < HIDDEN / 32) {  // hint next K tiles into cache
      __builtin_prefetch(A + (size_t)(m0 + (tid >> 1)) * HIDDEN + k0 + 32, 0, 0);
      __builtin_prefetch(W + (size_t)(k0 + 32 + (tid & 31)) * HIDDEN + n0, 0, 0);
    }
    __syncthreads();

    bf16x16 af[2], bfr[4];
#pragma unroll
    for (int i = 0; i < 2; ++i)
      af[i] = ldsA_frag(sA + (wm * 32 + i * 16) * 32, 32);
#pragma unroll
    for (int j = 0; j < 4; ++j)
      bfr[j] = ldsBT_frag(sBT + (wn * 64 + j * 16) * 32, 32);
#pragma unroll
    for (int i = 0; i < 2; ++i)
#pragma unroll
      for (int j = 0; j < 4; ++j)
        acc[i][j] = wmma_bf16(af[i], bfr[j], acc[i][j]);
    __syncthreads();
  }

  // Epilogue: +bias, write bf16 or f32.  D layout: col=lane&15, M=r+8*(lane>=16)
  const int nl = lane & 15, mb = (lane >> 4) * 8;
#pragma unroll
  for (int j = 0; j < 4; ++j) {
    const int ng = n0 + wn * 64 + j * 16 + nl;
    const float bv = bias[ng];
#pragma unroll
    for (int i = 0; i < 2; ++i) {
#pragma unroll
      for (int r = 0; r < 8; ++r) {
        const int mg = m0 + wm * 32 + i * 16 + mb + r;
        const float val = acc[i][j][r] + bv;
        if (outB) outB[(size_t)mg * HIDDEN + ng] = (bf16)val;
        else      outF[(size_t)mg * HIDDEN + ng] = val;
      }
    }
  }
}

// ---------------------------------------------------------------------------
// Ranking shortcut: importance[h][m] = scale * dot(mean_n q[n], k[m]).
// (mean over queries commutes with the dot product, so the reference's dense
//  QK^T ranking pass collapses to one mean-query vector per head.)
// ---------------------------------------------------------------------------
__global__ __launch_bounds__(128)
void meanq_kernel(const bf16* __restrict__ Qb, float* __restrict__ qmean) {
  const int h = blockIdx.x, d = threadIdx.x;
  const bf16* p = Qb + h * HEAD_DIM + d;
  float s = 0.0f;
  for (int n = 0; n < NSEQ; ++n) s += (float)p[(size_t)n * HIDDEN];
  qmean[h * HEAD_DIM + d] = s * (1.0f / (float)NSEQ);
}

__global__ __launch_bounds__(256)
void importance_kernel(const bf16* __restrict__ Kb, const float* __restrict__ qmean,
                       float* __restrict__ imp) {
  __shared__ float sq[HEAD_DIM];
  const int h = blockIdx.x;
  if (threadIdx.x < HEAD_DIM) sq[threadIdx.x] = qmean[h * HEAD_DIM + threadIdx.x];
  __syncthreads();
  const int m = blockIdx.y * 256 + threadIdx.x;
  const bf16* kr = Kb + (size_t)m * HIDDEN + h * HEAD_DIM;
  float dot = 0.0f;
#pragma unroll
  for (int c = 0; c < 16; ++c) {
    union { uint4 u; bf16 hh[8]; } x;
    x.u = *reinterpret_cast<const uint4*>(kr + c * 8);
#pragma unroll
    for (int i = 0; i < 8; ++i) dot += (float)x.hh[i] * sq[c * 8 + i];
  }
  imp[h * NSEQ + m] = dot * 0.08838834764831845f;  // 1/sqrt(128)
}

// ---------------------------------------------------------------------------
// Per-head top-256 via LDS bitonic sort (descending, lower index wins ties).
// ---------------------------------------------------------------------------
__global__ __launch_bounds__(256)
void topk_kernel(const float* __restrict__ imp, int* __restrict__ idx) {
  __shared__ float val[NSEQ];
  __shared__ int   ind[NSEQ];
  const int h = blockIdx.x, tid = threadIdx.x;
  for (int j = tid; j < NSEQ; j += 256) { val[j] = imp[h * NSEQ + j]; ind[j] = j; }
  for (unsigned k = 2; k <= (unsigned)NSEQ; k <<= 1) {
    for (unsigned s = k >> 1; s > 0; s >>= 1) {
      __syncthreads();
      for (unsigned j = tid; j < (unsigned)NSEQ; j += 256) {
        const unsigned x = j ^ s;
        if (x > j) {
          const float vj = val[j], vx = val[x];
          const int   ij = ind[j], ix = ind[x];
          const bool jFirst = (vj > vx) || (vj == vx && ij < ix);
          const bool desc = ((j & k) == 0);
          if (desc ? !jFirst : jFirst) {
            val[j] = vx; val[x] = vj; ind[j] = ix; ind[x] = ij;
          }
        }
      }
    }
  }
  __syncthreads();
  idx[h * TOPK + tid] = ind[tid];  // 256 threads == TOPK
}

// ---------------------------------------------------------------------------
// Gather Ksel[h][j][d] and Vsel^T[h][d][j] (transposed for B-fragment loads).
// ---------------------------------------------------------------------------
__global__ __launch_bounds__(256)
void gather_kernel(const bf16* __restrict__ Kb, const bf16* __restrict__ Vb,
                   const int* __restrict__ idx,
                   bf16* __restrict__ ksel, bf16* __restrict__ vselT) {
  const int t = blockIdx.x * 256 + threadIdx.x;  // < HEADS*TOPK*HEAD_DIM
  const int h = t >> 15, rem = t & 32767;
  {
    const int j = rem >> 7, d = rem & 127;
    const int row = idx[h * TOPK + j];
    ksel[(size_t)h * (TOPK * HEAD_DIM) + j * HEAD_DIM + d] =
        Kb[(size_t)row * HIDDEN + h * HEAD_DIM + d];
  }
  {
    const int d = rem >> 8, j = rem & 255;
    const int row = idx[h * TOPK + j];
    vselT[(size_t)h * (TOPK * HEAD_DIM) + d * TOPK + j] =
        Vb[(size_t)row * HIDDEN + h * HEAD_DIM + d];
  }
}

// ---------------------------------------------------------------------------
// Sparse attention: per (head, 64-query tile).
//   S(64x256) = Qtile(64x128) * Ksel^T     (WMMA, K-loop over d)
//   P = softmax_row(S * scale)             (f32, -> bf16)
//   O(64x128) = P * Vsel                   (WMMA, K-loop over j)
// Dynamic LDS: sQ 16K | sK 64K | sVT 64K | sS 64K = 208 KiB; sP reuses sQ/sK.
// Tile staging goes through the Tensor Data Mover (TENSORcnt), issued once by
// wave 0 (TDM ignores EXEC; one issue per wave avoids duplicate DMAs).
// ---------------------------------------------------------------------------
__global__ __launch_bounds__(256)
void sparse_attn(const bf16* __restrict__ Qb, const bf16* __restrict__ ksel,
                 const bf16* __restrict__ vselT, float* __restrict__ ctx) {
  extern __shared__ char smem[];
  bf16*  sQ  = (bf16*)smem;                   // 64 x 128
  bf16*  sK  = (bf16*)(smem + 16 * 1024);     // 256 x 128  (== B^T for S-GEMM)
  bf16*  sVT = (bf16*)(smem + 80 * 1024);     // 128 x 256  (== B^T for O-GEMM)
  float* sS  = (float*)(smem + 144 * 1024);   // 64 x 256
  bf16*  sP  = (bf16*)smem;                   // 64 x 256, reuses sQ/sK region

  const int tid = threadIdx.x, lane = tid & 31, wave = tid >> 5;
  const int h = blockIdx.x, q0 = blockIdx.y * 64;
  const float scale = 0.08838834764831845f;

  const bf16* kh = ksel + (size_t)h * (TOPK * HEAD_DIM);
  const bf16* vh = vselT + (size_t)h * (TOPK * HEAD_DIM);

#if USE_TDM
  if (wave == 0) {
    const unsigned ldsbase = __builtin_amdgcn_groupstaticsize();
    // sQ: strided 2D tile, 64 rows x 128 bf16, row stride HIDDEN elements.
    tdm_load_2d_b16(ldsbase + 0u,
                    Qb + (size_t)q0 * HIDDEN + h * HEAD_DIM,
                    /*tile*/ 128u, 64u, /*tensor*/ 128u, 64u,
                    /*stride0*/ (unsigned long long)HIDDEN);
    // sK / sVT: flat 32768-element (64 KiB) copies.
    tdm_load_2d_b16(ldsbase + 16u * 1024u, kh, 32768u, 1u, 32768u, 1u, 32768ull);
    tdm_load_2d_b16(ldsbase + 80u * 1024u, vh, 32768u, 1u, 32768u, 1u, 32768ull);
    __builtin_amdgcn_s_wait_tensorcnt(0);
  }
#else
  for (int l = tid; l < 1024; l += 256) {  // Q tile
    const int row = l >> 4, c8 = l & 15;
    *reinterpret_cast<uint4*>(&sQ[row * 128 + c8 * 8]) =
        *reinterpret_cast<const uint4*>(
            Qb + (size_t)(q0 + row) * HIDDEN + h * HEAD_DIM + c8 * 8);
  }
  for (int l = tid; l < 4096; l += 256)
    *reinterpret_cast<uint4*>(&sK[l * 8]) =
        *reinterpret_cast<const uint4*>(kh + l * 8);
  for (int l = tid; l < 4096; l += 256)
    *reinterpret_cast<uint4*>(&sVT[l * 8]) =
        *reinterpret_cast<const uint4*>(vh + l * 8);
#endif
  __syncthreads();

  {  // S = Q * Ksel^T : waves = 4(m of 16) x 2(j of 128)
    const int wmq = wave & 3, wnj = wave >> 2;
    f32x8 acc[8];
#pragma unroll
    for (int j = 0; j < 8; ++j) acc[j] = zero8();
    for (int kt = 0; kt < 4; ++kt) {
      bf16x16 a = ldsA_frag(sQ + (wmq * 16) * 128 + kt * 32, 128);
#pragma unroll
      for (int j = 0; j < 8; ++j) {
        bf16x16 b = ldsBT_frag(sK + (wnj * 128 + j * 16) * 128 + kt * 32, 128);
        acc[j] = wmma_bf16(a, b, acc[j]);
      }
    }
    const int nl = lane & 15, mb = (lane >> 4) * 8;
#pragma unroll
    for (int j = 0; j < 8; ++j)
#pragma unroll
      for (int r = 0; r < 8; ++r)
        sS[(wmq * 16 + mb + r) * 256 + wnj * 128 + j * 16 + nl] =
            acc[j][r] * scale;
  }
  __syncthreads();

  if (tid < 64) {  // row softmax -> bf16 P
    const float* srow = sS + tid * 256;
    float m = -3.4e38f;
    for (int j = 0; j < 256; ++j) m = fmaxf(m, srow[j]);
    float sum = 0.0f;
    for (int j = 0; j < 256; ++j) sum += __expf(srow[j] - m);
    const float inv = 1.0f / sum;
    bf16* prow = sP + tid * 256;
    for (int j = 0; j < 256; ++j) prow[j] = (bf16)(__expf(srow[j] - m) * inv);
  }
  __syncthreads();

  {  // O = P * Vsel : waves = 4(m of 16) x 2(d of 64)
    const int wmq = wave & 3, wnd = wave >> 2;
    f32x8 acc[4];
#pragma unroll
    for (int j = 0; j < 4; ++j) acc[j] = zero8();
    for (int kt = 0; kt < 8; ++kt) {
      bf16x16 a = ldsA_frag(sP + (wmq * 16) * 256 + kt * 32, 256);
#pragma unroll
      for (int j = 0; j < 4; ++j) {
        bf16x16 b = ldsBT_frag(sVT + (wnd * 64 + j * 16) * 256 + kt * 32, 256);
        acc[j] = wmma_bf16(a, b, acc[j]);
      }
    }
    const int nl = lane & 15, mb = (lane >> 4) * 8;
#pragma unroll
    for (int j = 0; j < 4; ++j)
#pragma unroll
      for (int r = 0; r < 8; ++r)
        ctx[(size_t)(q0 + wmq * 16 + mb + r) * HIDDEN + h * HEAD_DIM +
            wnd * 64 + j * 16 + nl] = acc[j][r];
  }
}

// ---------------------------------------------------------------------------
extern "C" void kernel_launch(void* const* d_in, const int* in_sizes, int n_in,
                              void* d_out, int out_size, void* d_ws, size_t ws_size,
                              hipStream_t stream) {
  (void)in_sizes; (void)n_in; (void)out_size; (void)ws_size;
  const float* X  = (const float*)d_in[0];
  const float* Wq = (const float*)d_in[1];
  const float* bq = (const float*)d_in[2];
  const float* Wk = (const float*)d_in[3];
  const float* bk = (const float*)d_in[4];
  const float* Wv = (const float*)d_in[5];
  const float* bv = (const float*)d_in[6];
  const float* Wo = (const float*)d_in[7];
  const float* bo = (const float*)d_in[8];
  float* out = (float*)d_out;

  // Workspace layout (≈43 MiB total)
  char* ws = (char*)d_ws;
  bf16*  Qb    = (bf16*)(ws + (size_t)0);
  bf16*  Kb    = (bf16*)(ws + ((size_t)8 << 20));
  bf16*  Vb    = (bf16*)(ws + ((size_t)16 << 20));
  float* ctx   = (float*)(ws + ((size_t)24 << 20));                // 16 MiB
  float* qmean = (float*)(ws + ((size_t)40 << 20));                // 8 KiB
  float* imp   = (float*)(ws + ((size_t)40 << 20) + (16 << 10));   // 128 KiB
  int*   idx   = (int*)  (ws + ((size_t)40 << 20) + (144 << 10));  // 16 KiB
  bf16*  ksel  = (bf16*)(ws + ((size_t)41 << 20));                 // 1 MiB
  bf16*  vselT = (bf16*)(ws + ((size_t)42 << 20));                 // 1 MiB

  const dim3 gg(16, 16);
  gemm2048_bf16wmma<<<gg, 256, 0, stream>>>(X, Wq, bq, Qb, nullptr);
  gemm2048_bf16wmma<<<gg, 256, 0, stream>>>(X, Wk, bk, Kb, nullptr);
  gemm2048_bf16wmma<<<gg, 256, 0, stream>>>(X, Wv, bv, Vb, nullptr);

  meanq_kernel<<<16, 128, 0, stream>>>(Qb, qmean);
  importance_kernel<<<dim3(16, 8), 256, 0, stream>>>(Kb, qmean, imp);
  topk_kernel<<<16, 256, 0, stream>>>(imp, idx);
  gather_kernel<<<2048, 256, 0, stream>>>(Kb, Vb, idx, ksel, vselT);

  const size_t smem_bytes = 208 * 1024;  // <= 320 KiB/workgroup on CDNA5
  (void)hipFuncSetAttribute((const void*)sparse_attn,
                            hipFuncAttributeMaxDynamicSharedMemorySize,
                            (int)smem_bytes);
  sparse_attn<<<dim3(16, 32), 256, smem_bytes, stream>>>(Qb, ksel, vselT, ctx);

  gemm2048_bf16wmma<<<gg, 256, 0, stream>>>(ctx, Wo, bo, nullptr, out);
}
